// TransformerBlock_90872918048899
// MI455X (gfx1250) — matrix operand order repaired and checked by reading, not compiled, not run on hardware
//
#include <hip/hip_runtime.h>

// ---------------------------------------------------------------------------
// TransformerBlock (attention + top2-MoE) for gfx1250, f16 WMMA + f32 accum.
// B=4 T=2048 C=1024 H=16 d=64 E=8 TOPK=2, N = B*T = 8192 tokens.
// Projection GEMMs: 4 waves/block share a 64x32 B-panel staged into LDS via
// async global->LDS copies (ASYNCcnt), double-buffered; 32x64 tile per wave.
// ---------------------------------------------------------------------------

typedef _Float16 v8h  __attribute__((ext_vector_type(8)));
typedef _Float16 v16h __attribute__((ext_vector_type(16)));
typedef float    v8f  __attribute__((ext_vector_type(8)));

#define NTOK 8192
#define CDIM 1024
#define TDIM 2048
#define HEADS 16
#define DHEAD 64
#define NEXP 8

__device__ __forceinline__ v8h ld8h(const _Float16* p) { return *(const v8h*)p; }

__device__ __forceinline__ v16h cat16(v8h lo, v8h hi) {
  return __builtin_shufflevector(lo, hi, 0,1,2,3,4,5,6,7,8,9,10,11,12,13,14,15);
}

// A-matrix (16xK tile, row-major source, f16). Documented CDNA5 layout:
// lanes 0-15 hold row m, halfs = K kb..kb+7 then kb+16..kb+23;
// lanes 16-31 hold row m, halfs = K kb+8..kb+15 then kb+24..kb+31.
__device__ __forceinline__ v16h load_A(const _Float16* M, int ld, int rm, int kb, int lane) {
  int r  = rm + (lane & 15);
  int k0 = kb + ((lane >> 4) << 3);   // +0 or +8
  const _Float16* p = M + (size_t)r * ld + k0;
  return cat16(ld8h(p), ld8h(p + 16));
}

// B-matrix (Kx16 tile). Column n of B = row n of W (row-major, K contiguous):
// lanes 0-15: column n, K kb..kb+7 then kb+16..kb+23; lanes 16-31: column n, K kb+8..kb+15 then kb+24..kb+31.
__device__ __forceinline__ v16h load_B(const _Float16* W, int ld, int cn, int kb, int lane) {
  int n  = cn + (lane & 15);
  int k0 = kb + ((lane >> 4) << 3);   // +0 or +8
  const _Float16* p = W + (size_t)n * ld + k0;
  return cat16(ld8h(p), ld8h(p + 16));
}

__device__ __forceinline__ v8f wmma16(v16h a, v16h b, v8f c) {
  return __builtin_amdgcn_wmma_f32_16x16x32_f16(false, a, false, b, (short)0, c,
                                                false, false);
}

// Async copy of 16B/lane from global to LDS (ASYNCcnt-tracked, ISA 15.18.3).
__device__ __forceinline__ void async_b128(unsigned lds_byte_off, const void* g) {
  asm volatile("global_load_async_to_lds_b128 %0, %1, off"
               :: "v"(lds_byte_off), "v"((unsigned long long)(size_t)g)
               : "memory");
}
__device__ __forceinline__ void wait_async0() {
  asm volatile("s_wait_asynccnt 0x0" ::: "memory");
}

// ---------------------------------------------------------------------------
__global__ void k_cast_f16(const float* __restrict__ src, _Float16* __restrict__ dst, int n) {
  int i = blockIdx.x * blockDim.x + threadIdx.x;
  int st = gridDim.x * blockDim.x;
  for (; i < n; i += st) dst[i] = (_Float16)src[i];
}

// One block per token.
__global__ __launch_bounds__(256)
void k_rmsnorm(const float* __restrict__ x, const float* __restrict__ w,
               _Float16* __restrict__ out) {
  int n = blockIdx.x;
  const float* xr = x + (size_t)n * CDIM;
  float s = 0.f;
  for (int c = threadIdx.x; c < CDIM; c += 256) { float v = xr[c]; s += v * v; }
  __shared__ float red[8];
  for (int o = 16; o > 0; o >>= 1) s += __shfl_down(s, o, 32);
  if ((threadIdx.x & 31) == 0) red[threadIdx.x >> 5] = s;
  __syncthreads();
  if (threadIdx.x < 8) {
    float t = red[threadIdx.x];
    for (int o = 4; o > 0; o >>= 1) t += __shfl_down(t, o, 8);
    if (threadIdx.x == 0) red[0] = t;
  }
  __syncthreads();
  float rs = rsqrtf(red[0] * (1.f / CDIM) + 1e-6f);
  for (int c = threadIdx.x; c < CDIM; c += 256)
    out[(size_t)n * CDIM + c] = (_Float16)(xr[c] * rs * w[c]);
}

// out = A @ W^T + pos, stored f16. vmode=0: (b,h,t,d); vmode=1: (b,h,d,t).
// Block: 4 waves x 32 rows, shared 64-col B panel double-buffered in LDS.
__global__ __launch_bounds__(128)
void k_gemm_qkv(const _Float16* __restrict__ A, const _Float16* __restrict__ W,
                const float* __restrict__ pos, _Float16* __restrict__ out, int vmode) {
  const int lane = threadIdx.x & 31;
  const int wid  = threadIdx.x >> 5;
  const int rm = (blockIdx.x >> 4) * 128 + wid * 32;   // per-wave 32-row tile
  const int cn = (blockIdx.x & 15) << 6;               // shared 64-col panel

  __shared__ __align__(16) _Float16 bbuf[2][64][32];   // [buf][panel row][k]

  const char* wb = (const char*)W;
  auto issueB = [&](int buf, int kb) {                 // wave 0 only
    unsigned base = (unsigned)(size_t)&bbuf[buf][0][0];
#pragma unroll
    for (int i = 0; i < 8; ++i) {
      unsigned o   = i * 512 + lane * 16;              // 4KB panel, 16B/lane
      unsigned row = o >> 6;                           // 64B per panel row
      unsigned kby = o & 63;
      async_b128(base + o, wb + ((size_t)(cn + row) * CDIM + kb) * 2 + kby);
    }
  };

  if (wid == 0) issueB(0, 0);
  v8f acc0[4] = {}, acc1[4] = {};
  int buf = 0;
  for (int kb = 0; kb < CDIM; kb += 32, buf ^= 1) {
    if (wid == 0) wait_async0();
    __syncthreads();                                   // panel[buf] ready
    if (wid == 0 && kb + 32 < CDIM) issueB(buf ^ 1, kb + 32);
    v16h a0 = load_A(A, CDIM, rm,      kb, lane);
    v16h a1 = load_A(A, CDIM, rm + 16, kb, lane);
#pragma unroll
    for (int sub = 0; sub < 4; ++sub) {
      const _Float16* bp = &bbuf[buf][sub * 16 + (lane & 15)][(lane >> 4) << 3];
      v16h bw = cat16(ld8h(bp), ld8h(bp + 16));
      acc0[sub] = wmma16(a0, bw, acc0[sub]);
      acc1[sub] = wmma16(a1, bw, acc1[sub]);
    }
    __syncthreads();                                   // all done with panel[buf]
  }
  int ncol = lane & 15, mbase = (lane >> 4) << 3;
#pragma unroll
  for (int set = 0; set < 2; ++set) {
    const v8f* acc = set ? acc1 : acc0;
#pragma unroll
    for (int sub = 0; sub < 4; ++sub) {
      int j = cn + sub * 16 + ncol;
      int hh = j >> 6, dc = j & 63;
#pragma unroll
      for (int r = 0; r < 8; ++r) {
        int ntok = rm + set * 16 + mbase + r;
        int bb = ntok >> 11, tt = ntok & (TDIM - 1);
        float val = acc[sub][r] + pos[(size_t)tt * CDIM + j];
        size_t o = vmode ? ((((size_t)bb * HEADS + hh) * DHEAD + dc) * TDIM + tt)
                         : ((((size_t)bb * HEADS + hh) * TDIM + tt) * DHEAD + dc);
        out[o] = (_Float16)val;
      }
    }
  }
}

// h = x + y @ Wo^T (f32), mirrored into d_out as the MoE residual base.
__global__ __launch_bounds__(128)
void k_gemm_oproj(const _Float16* __restrict__ A, const _Float16* __restrict__ W,
                  const float* __restrict__ x, float* __restrict__ hout,
                  float* __restrict__ outdup) {
  const int lane = threadIdx.x & 31;
  const int wid  = threadIdx.x >> 5;
  const int rm = (blockIdx.x >> 4) * 128 + wid * 32;
  const int cn = (blockIdx.x & 15) << 6;

  __shared__ __align__(16) _Float16 bbuf[2][64][32];
  const char* wb = (const char*)W;
  auto issueB = [&](int buf, int kb) {
    unsigned base = (unsigned)(size_t)&bbuf[buf][0][0];
#pragma unroll
    for (int i = 0; i < 8; ++i) {
      unsigned o   = i * 512 + lane * 16;
      unsigned row = o >> 6;
      unsigned kby = o & 63;
      async_b128(base + o, wb + ((size_t)(cn + row) * CDIM + kb) * 2 + kby);
    }
  };

  if (wid == 0) issueB(0, 0);
  v8f acc0[4] = {}, acc1[4] = {};
  int buf = 0;
  for (int kb = 0; kb < CDIM; kb += 32, buf ^= 1) {
    if (wid == 0) wait_async0();
    __syncthreads();
    if (wid == 0 && kb + 32 < CDIM) issueB(buf ^ 1, kb + 32);
    v16h a0 = load_A(A, CDIM, rm,      kb, lane);
    v16h a1 = load_A(A, CDIM, rm + 16, kb, lane);
#pragma unroll
    for (int sub = 0; sub < 4; ++sub) {
      const _Float16* bp = &bbuf[buf][sub * 16 + (lane & 15)][(lane >> 4) << 3];
      v16h bw = cat16(ld8h(bp), ld8h(bp + 16));
      acc0[sub] = wmma16(a0, bw, acc0[sub]);
      acc1[sub] = wmma16(a1, bw, acc1[sub]);
    }
    __syncthreads();
  }
  int ncol = lane & 15, mbase = (lane >> 4) << 3;
#pragma unroll
  for (int set = 0; set < 2; ++set) {
    const v8f* acc = set ? acc1 : acc0;
#pragma unroll
    for (int sub = 0; sub < 4; ++sub) {
      int j = cn + sub * 16 + ncol;
#pragma unroll
      for (int r = 0; r < 8; ++r) {
        int ntok = rm + set * 16 + mbase + r;
        size_t o = (size_t)ntok * CDIM + j;
        float val = x[o] + acc[sub][r];
        hout[o] = val;
        outdup[o] = val;
      }
    }
  }
}

// Flash attention: 4 independent waves per block, 1 wave per (b, h, 16-query
// tile), per-wave LDS slice (no block barrier). scale = 1/sqrt(C) = 1/32.
__global__ __launch_bounds__(128)
void k_attn(const _Float16* __restrict__ q, const _Float16* __restrict__ k,
            const _Float16* __restrict__ vT, _Float16* __restrict__ y) {
  const int lane = threadIdx.x & 31;
  const int wid  = threadIdx.x >> 5;
  const int hh = blockIdx.y, bb = blockIdx.z;
  const int q0 = (blockIdx.x * 4 + wid) << 4;
  const size_t bh = (size_t)bb * HEADS + hh;
  const _Float16* qp = q  + bh * (size_t)TDIM * DHEAD;
  const _Float16* kp = k  + bh * (size_t)TDIM * DHEAD;
  const _Float16* vp = vT + bh * (size_t)DHEAD * TDIM;

  v16h aq0 = load_A(qp, DHEAD, q0, 0, lane);
  v16h aq1 = load_A(qp, DHEAD, q0, 32, lane);

  v8f acc[4] = {};
  float mrow[8], lrow[8];
#pragma unroll
  for (int r = 0; r < 8; ++r) { mrow[r] = -__builtin_inff(); lrow[r] = 0.f; }

  __shared__ __align__(16) _Float16 pbuf[4][16][32];

  const int ncol = lane & 15, mbase = (lane >> 4) << 3;
  const float scale = 0.03125f;   // 1/sqrt(1024)

  for (int s0 = 0; s0 <= q0 + 15; s0 += 32) {
    v8f sa = {}, sb = {};
    { v16h bk = load_B(kp, DHEAD, s0, 0, lane);       sa = wmma16(aq0, bk, sa);
      bk      = load_B(kp, DHEAD, s0, 32, lane);      sa = wmma16(aq1, bk, sa); }
    { v16h bk = load_B(kp, DHEAD, s0 + 16, 0, lane);  sb = wmma16(aq0, bk, sb);
      bk      = load_B(kp, DHEAD, s0 + 16, 32, lane); sb = wmma16(aq1, bk, sb); }

#pragma unroll
    for (int r = 0; r < 8; ++r) {
      int qm = q0 + mbase + r;
      float va = (s0 + ncol      <= qm) ? sa[r] * scale : -__builtin_inff();
      float vb = (s0 + 16 + ncol <= qm) ? sb[r] * scale : -__builtin_inff();
      float mx = fmaxf(va, vb);
      for (int o = 8; o; o >>= 1) mx = fmaxf(mx, __shfl_xor(mx, o, 32));
      float mnew = fmaxf(mrow[r], mx);
      float corr = __expf(mrow[r] - mnew);
      float pa = __expf(va - mnew);
      float pb = __expf(vb - mnew);
      sa[r] = pa; sb[r] = pb;
      float sm = pa + pb;
      for (int o = 8; o; o >>= 1) sm += __shfl_xor(sm, o, 32);
      lrow[r] = lrow[r] * corr + sm;
      mrow[r] = mnew;
      acc[0][r] *= corr; acc[1][r] *= corr; acc[2][r] *= corr; acc[3][r] *= corr;
    }
    // Stage P (C-layout) into per-wave LDS slice, read back in A-layout.
#pragma unroll
    for (int r = 0; r < 8; ++r) {
      int m = mbase + r;
      pbuf[wid][m][ncol]      = (_Float16)sa[r];
      pbuf[wid][m][16 + ncol] = (_Float16)sb[r];
    }
    asm volatile("s_wait_dscnt 0" ::: "memory");
    v16h ap;
    {
      int m = lane & 15;
      int klo = (lane >> 4) ? 8 : 0;
      ap = cat16(*(const v8h*)&pbuf[wid][m][klo], *(const v8h*)&pbuf[wid][m][klo + 16]);
    }
#pragma unroll
    for (int sub = 0; sub < 4; ++sub) {
      int dn = sub * 16 + ncol;
      int kk = s0 + ((lane >> 4) << 3);
      const _Float16* pv = vp + (size_t)dn * TDIM + kk;
      v16h bv = cat16(ld8h(pv), ld8h(pv + 16));
      acc[sub] = wmma16(ap, bv, acc[sub]);
    }
  }
#pragma unroll
  for (int sub = 0; sub < 4; ++sub)
#pragma unroll
    for (int r = 0; r < 8; ++r) {
      int m = mbase + r;
      float o = acc[sub][r] / lrow[r];
      y[((size_t)bb * TDIM + q0 + m) * CDIM + hh * DHEAD + sub * 16 + ncol] = (_Float16)o;
    }
}

__global__ void k_zero_counts(int* counts) { if (threadIdx.x < NEXP) counts[threadIdx.x] = 0; }

// 8 waves / block, one token per wave: gate logits, top-2 softmax, counts.
__global__ __launch_bounds__(256)
void k_gate(const _Float16* __restrict__ hn, const float* __restrict__ gw,
            int* __restrict__ idx, float* __restrict__ wgt, int* __restrict__ counts) {
  int lane = threadIdx.x & 31;
  int n = blockIdx.x * 8 + (threadIdx.x >> 5);
  const _Float16* xr = hn + (size_t)n * CDIM;
  float lg[NEXP] = {};
  for (int c = lane; c < CDIM; c += 32) {
    float xv = (float)xr[c];
#pragma unroll
    for (int e = 0; e < NEXP; ++e) lg[e] += xv * gw[e * CDIM + c];
  }
#pragma unroll
  for (int e = 0; e < NEXP; ++e)
    for (int o = 16; o; o >>= 1) lg[e] += __shfl_xor(lg[e], o, 32);
  if (lane == 0) {
    int i0 = 0;
    for (int e = 1; e < NEXP; ++e) if (lg[e] > lg[i0]) i0 = e;
    int i1 = (i0 == 0) ? 1 : 0;
    for (int e = 0; e < NEXP; ++e) if (e != i0 && lg[e] > lg[i1]) i1 = e;
    float e1 = __expf(lg[i1] - lg[i0]);
    float inv = 1.f / (1.f + e1);
    idx[2 * n] = i0; idx[2 * n + 1] = i1;
    wgt[2 * n] = inv; wgt[2 * n + 1] = e1 * inv;
    atomicAdd(&counts[i0], 1);
    atomicAdd(&counts[i1], 1);
  }
}

__global__ void k_offsets(const int* counts, int* offs, int* cursor) {
  if (threadIdx.x == 0 && blockIdx.x == 0) {
    int s = 0;
    for (int e = 0; e < NEXP; ++e) { offs[e] = s; cursor[e] = s; s += counts[e]; }
  }
}

__global__ void k_scatter(const int* __restrict__ idx, int* cursor,
                          int* __restrict__ list, int total) {
  int i = blockIdx.x * blockDim.x + threadIdx.x;
  if (i < total) {
    int pos = atomicAdd(&cursor[idx[i]], 1);
    list[pos] = i;                          // encodes token (i>>1) and slot (i&1)
  }
}

// Per-expert gathered GEMM, 32-token x 64-col wave tile:
// out[tok, :] += w * (hn[tok, :] @ We^T).
__global__ __launch_bounds__(32)
void k_moe_gemm(const _Float16* __restrict__ hn, const _Float16* __restrict__ ew16,
                const int* __restrict__ list, const int* __restrict__ offs,
                const int* __restrict__ counts, const float* __restrict__ wgt,
                float* __restrict__ out) {
  int e = blockIdx.y;
  int cnt = counts[e];
  int tm = blockIdx.x >> 4;
  int tn = blockIdx.x & 15;
  if (tm * 32 >= cnt) return;
  const int lane = threadIdx.x;
  const int* lst = list + offs[e];
  int row0 = tm * 32 + (lane & 15);
  int row1 = row0 + 16;
  int tok0 = lst[(row0 < cnt) ? row0 : (cnt - 1)] >> 1;
  int tok1 = lst[(row1 < cnt) ? row1 : (cnt - 1)] >> 1;
  const _Float16* W = ew16 + (size_t)e * CDIM * CDIM;
  int cn = tn << 6;
  v8f acc0[4] = {}, acc1[4] = {};
  for (int kb = 0; kb < CDIM; kb += 32) {
    int k0 = kb + ((lane >> 4) << 3);
    const _Float16* pa0 = hn + (size_t)tok0 * CDIM + k0;  // gathered A rows
    const _Float16* pa1 = hn + (size_t)tok1 * CDIM + k0;
    v16h a0 = cat16(ld8h(pa0), ld8h(pa0 + 16));
    v16h a1 = cat16(ld8h(pa1), ld8h(pa1 + 16));
#pragma unroll
    for (int sub = 0; sub < 4; ++sub) {
      v16h bw = load_B(W, CDIM, cn + sub * 16, kb, lane);
      acc0[sub] = wmma16(a0, bw, acc0[sub]);
      acc1[sub] = wmma16(a1, bw, acc1[sub]);
    }
  }
  int ncol = lane & 15, mbase = (lane >> 4) << 3;
#pragma unroll
  for (int set = 0; set < 2; ++set) {
    const v8f* acc = set ? acc1 : acc0;
#pragma unroll
    for (int sub = 0; sub < 4; ++sub) {
      int j = cn + sub * 16 + ncol;
#pragma unroll
      for (int r = 0; r < 8; ++r) {
        int rowi = tm * 32 + set * 16 + mbase + r;
        if (rowi < cnt) {
          int en = lst[rowi];
          atomicAdd(&out[(size_t)(en >> 1) * CDIM + j], wgt[en] * acc[sub][r]);
        }
      }
    }
  }
}

// ---------------------------------------------------------------------------
extern "C" void kernel_launch(void* const* d_in, const int* in_sizes, int n_in,
                              void* d_out, int out_size, void* d_ws, size_t ws_size,
                              hipStream_t stream) {
  (void)in_sizes; (void)n_in; (void)out_size; (void)ws_size;
  const float* x   = (const float*)d_in[0];
  const float* pos = (const float*)d_in[1];
  const float* wq  = (const float*)d_in[2];
  const float* wk  = (const float*)d_in[3];
  const float* wv  = (const float*)d_in[4];
  const float* wo  = (const float*)d_in[5];
  const float* gw  = (const float*)d_in[6];
  const float* ew  = (const float*)d_in[7];
  const float* anw = (const float*)d_in[8];
  const float* fnw = (const float*)d_in[9];
  float* out = (float*)d_out;

  char* p = (char*)d_ws;
  auto alloc = [&](size_t bytes) -> void* {
    void* r = (void*)p; p += (bytes + 255) & ~(size_t)255; return r;
  };
  const size_t CC = (size_t)CDIM * CDIM;
  _Float16* wq16 = (_Float16*)alloc(CC * 2);
  _Float16* wk16 = (_Float16*)alloc(CC * 2);
  _Float16* wv16 = (_Float16*)alloc(CC * 2);
  _Float16* wo16 = (_Float16*)alloc(CC * 2);
  _Float16* ew16 = (_Float16*)alloc(NEXP * CC * 2);
  _Float16* xn   = (_Float16*)alloc((size_t)NTOK * CDIM * 2);
  _Float16* qf   = (_Float16*)alloc((size_t)NTOK * CDIM * 2);
  _Float16* kf   = (_Float16*)alloc((size_t)NTOK * CDIM * 2);
  _Float16* vT   = (_Float16*)alloc((size_t)NTOK * CDIM * 2);
  _Float16* yf   = (_Float16*)alloc((size_t)NTOK * CDIM * 2);
  float*    hbuf = (float*)alloc((size_t)NTOK * CDIM * 4);
  _Float16* hn   = (_Float16*)alloc((size_t)NTOK * CDIM * 2);
  int*   idx     = (int*)alloc(2 * NTOK * 4);
  float* wgt     = (float*)alloc(2 * NTOK * 4);
  int*   counts  = (int*)alloc(NEXP * 4);
  int*   offs    = (int*)alloc(NEXP * 4);
  int*   cursor  = (int*)alloc(NEXP * 4);
  int*   list    = (int*)alloc(2 * NTOK * 4);

  // 1) weights -> f16
  k_cast_f16<<<1024, 256, 0, stream>>>(wq, wq16, (int)CC);
  k_cast_f16<<<1024, 256, 0, stream>>>(wk, wk16, (int)CC);
  k_cast_f16<<<1024, 256, 0, stream>>>(wv, wv16, (int)CC);
  k_cast_f16<<<1024, 256, 0, stream>>>(wo, wo16, (int)CC);
  k_cast_f16<<<4096, 256, 0, stream>>>(ew, ew16, (int)(NEXP * CC));

  // 2) rmsnorm(x) -> xn
  k_rmsnorm<<<NTOK, 256, 0, stream>>>(x, anw, xn);

  // 3) Q,K,V projections (+pos), V transposed per (b,h)
  k_gemm_qkv<<<1024, 128, 0, stream>>>(xn, wq16, pos, qf, 0);
  k_gemm_qkv<<<1024, 128, 0, stream>>>(xn, wk16, pos, kf, 0);
  k_gemm_qkv<<<1024, 128, 0, stream>>>(xn, wv16, pos, vT, 1);

  // 4) flash attention -> yf (4 waves/block, each wave one 16-query tile)
  k_attn<<<dim3(TDIM / 64, HEADS, NTOK / TDIM), 128, 0, stream>>>(qf, kf, vT, yf);

  // 5) h = x + yf @ wo^T  (also seeds d_out with h)
  k_gemm_oproj<<<1024, 128, 0, stream>>>(yf, wo16, x, hbuf, out);

  // 6) rmsnorm(h) -> hn; gating; build per-expert token lists
  k_rmsnorm<<<NTOK, 256, 0, stream>>>(hbuf, fnw, hn);
  k_zero_counts<<<1, 32, 0, stream>>>(counts);
  k_gate<<<NTOK / 8, 256, 0, stream>>>(hn, gw, idx, wgt, counts);
  k_offsets<<<1, 1, 0, stream>>>(counts, offs, cursor);
  k_scatter<<<(2 * NTOK + 255) / 256, 256, 0, stream>>>(idx, cursor, list, 2 * NTOK);

  // 7) top-2 expert GEMMs (32x64 wave tiles), scaled atomic accumulate
  k_moe_gemm<<<dim3((NTOK / 32) * (CDIM / 64), NEXP), 32, 0, stream>>>(
      hn, ew16, list, offs, counts, wgt, out);
}
